// Point2NeighborAttention_25314537242842
// MI455X (gfx1250) — matrix operand order repaired and checked by reading, not compiled, not run on hardware
//
#include <hip/hip_runtime.h>
#include <hip/hip_bf16.h>

// ---------------------------------------------------------------------------
// Point2NeighborAttention for MI455X (gfx1250, wave32, WMMA bf16 16x16x32)
//
// Algebraic restructuring:
//   logits = f^T (Wq^T Wk) g   -> precompute M = Wq^T Wk, qm = h*M  (WMMA)
//   x_r    = Wv (sum_j attn_j g_j) -> aggregate neighbors first     (VALU)
//   then u = s*Wv^T, t = h-u, ff = t*Wff^T fused in one kernel      (WMMA)
// Weight (B) matrices are pre-swizzled into WMMA fragment order so a
// B-fragment load is two global_load_b128; A fragments are two ds_load_b128.
// Whole activation set (8 MB) is L2-resident on the 192 MB L2.
// ---------------------------------------------------------------------------

#define BATCH   4
#define NPTS    4096
#define BN      16384      // BATCH*NPTS
#define CH      128
#define KNN_K   32
#define EPS_BN  1e-5f

typedef __attribute__((ext_vector_type(16))) __bf16 v16bf;
typedef __attribute__((ext_vector_type(8)))  float  v8f;

__device__ __forceinline__ unsigned short f2bf(float f) {
  union { float f; unsigned u; } v; v.f = f;
  unsigned r = v.u + 0x7FFFu + ((v.u >> 16) & 1u);   // round-to-nearest-even
  return (unsigned short)(r >> 16);
}

// ---------------------------------------------------------------------------
// WMMA fragment layouts per CDNA5 ISA 7.12.2 (wave32):
//  A 16x32 bf16: row = lane&15; halves 0-7 -> K = (lane>=16?8:0)+i,
//                halves 8-15 -> K = 16+(lane>=16?8:0)+(i-8)
//  B 32x16 bf16: col = lane&15; halves 0-7 -> K = (lane>=16?8:0)+h, 8-15 -> K = 16+(lane>=16?8:0)+(h-8)
//  C/D 16x16 f32: col = lane&15; row = vgpr + 8*(lane>>4)
//
// B-matrix swizzle: fragment halves stored contiguously per lane:
//  Bswz[(((cotile*4 + kb)*32 + lane) << 4) + h]
// ---------------------------------------------------------------------------
__device__ __forceinline__ int b_swz_index(int k, int o) {
  int cotile = o >> 4, colL = o & 15;
  int kb = k >> 5, kr = k & 31;
  int lane = (((kr >> 3) & 1) << 4) + colL; // ((kr>>3)&1 ? 16 : 0) + col
  int h = (kr & 7) + ((kr >> 4) << 3);
  return (((cotile * 4 + kb) * 32 + lane) << 4) + h;
}

__device__ __forceinline__ v16bf load_a_frag(const unsigned short* tile, int lane, int kb) {
  union { uint4 q[2]; v16bf v; } a;
  int row = lane & 15, hb = (lane >> 4) * 8;
  const unsigned short* p = tile + row * CH + kb * 32;
  a.q[0] = *reinterpret_cast<const uint4*>(p + hb);        // K = kb*32 + hb .. +7
  a.q[1] = *reinterpret_cast<const uint4*>(p + 16 + hb);   // K = kb*32+16+hb .. +7
  return a.v;
}

__device__ __forceinline__ v16bf load_b_frag_swz(const unsigned short* Bs, int lane,
                                                 int kb, int cotile) {
  union { uint4 q[2]; v16bf v; } b;
  const uint4* p = reinterpret_cast<const uint4*>(Bs + (((cotile * 4 + kb) * 32 + lane) << 4));
  b.q[0] = p[0];
  b.q[1] = p[1];
  return b.v;
}

// ---------------------------------------------------------------------------
// Kernel 1: fused MLP 3 -> 64 -> 128 with folded inference BN + ReLU.
// One wave per point; 8 points per 256-thread block.
// ---------------------------------------------------------------------------
__global__ void mlp_kernel(const float* __restrict__ x,
                           const float* __restrict__ W1, const float* __restrict__ W2,
                           const float* __restrict__ g1, const float* __restrict__ b1,
                           const float* __restrict__ m1, const float* __restrict__ v1,
                           const float* __restrict__ g2, const float* __restrict__ b2,
                           const float* __restrict__ m2, const float* __restrict__ v2,
                           float* __restrict__ h_f32, unsigned short* __restrict__ h_b16) {
  __shared__ __align__(16) float h1s[8][64];
  int tid = threadIdx.x, lane = tid & 31, wave = tid >> 5;
  int p = blockIdx.x * 8 + wave;
  const float* xp = x + (size_t)p * 3;
  float x0 = xp[0], x1 = xp[1], x2 = xp[2];
  #pragma unroll
  for (int t = 0; t < 2; ++t) {
    int o = lane * 2 + t;
    float a = W1[o * 3 + 0] * x0 + W1[o * 3 + 1] * x1 + W1[o * 3 + 2] * x2;
    float sc = g1[o] * rsqrtf(v1[o] + EPS_BN);
    h1s[wave][o] = fmaxf((a - m1[o]) * sc + b1[o], 0.0f);
  }
  __syncthreads();
  #pragma unroll
  for (int t = 0; t < 4; ++t) {
    int o = lane + t * 32;
    float acc = 0.0f;
    #pragma unroll
    for (int c = 0; c < 64; c += 4) {
      float4 w4 = *(const float4*)(W2 + o * 64 + c);
      float4 h4 = *(const float4*)&h1s[wave][c];
      acc += w4.x * h4.x + w4.y * h4.y + w4.z * h4.z + w4.w * h4.w;
    }
    float sc = g2[o] * rsqrtf(v2[o] + EPS_BN);
    acc = fmaxf((acc - m2[o]) * sc + b2[o], 0.0f);
    size_t off = (size_t)p * CH + o;
    h_f32[off] = acc;
    h_b16[off] = f2bf(acc);
  }
}

// ---------------------------------------------------------------------------
// Kernel 2: 32-NN. One wave per query; per-lane 32-entry candidate list in
// LDS, then 32 rounds of wave32 cross-lane argmin merge.
// ---------------------------------------------------------------------------
__global__ void knn_kernel(const float* __restrict__ x, int* __restrict__ idx_out) {
  __shared__ float sd[128 * KNN_K];
  __shared__ int   si[128 * KNN_K];
  int tid = threadIdx.x, lane = tid & 31, wave = tid >> 5;
  int q = blockIdx.x * 4 + wave;
  int b = q >> 12, n = q & (NPTS - 1);
  const float* xb = x + (size_t)b * NPTS * 3;
  float qx = xb[n * 3 + 0], qy = xb[n * 3 + 1], qz = xb[n * 3 + 2];
  float* md = sd + tid * KNN_K;
  int*   mi = si + tid * KNN_K;
  #pragma unroll
  for (int s = 0; s < KNN_K; ++s) { md[s] = 3.4e38f; mi[s] = 0; }
  float worst = 3.4e38f; int wslot = 0;
  for (int j = lane; j < NPTS; j += 32) {
    float dx = xb[j * 3 + 0] - qx, dy = xb[j * 3 + 1] - qy, dz = xb[j * 3 + 2] - qz;
    float d = dx * dx + dy * dy + dz * dz;
    if (d < worst) {
      md[wslot] = d; mi[wslot] = j;
      float w = -1.0f; int ws = 0;
      #pragma unroll
      for (int s = 0; s < KNN_K; ++s) { float v = md[s]; if (v > w) { w = v; ws = s; } }
      worst = w; wslot = ws;
    }
  }
  __syncthreads();
  int* out = idx_out + (size_t)q * KNN_K;
  for (int r = 0; r < KNN_K; ++r) {
    float lv = 3.4e38f; int ls = 0;
    #pragma unroll
    for (int s = 0; s < KNN_K; ++s) { float v = md[s]; if (v < lv) { lv = v; ls = s; } }
    float bv = lv; int bl = lane, bi = mi[ls];
    #pragma unroll
    for (int off = 16; off > 0; off >>= 1) {
      float ov = __shfl_xor(bv, off, 32);
      int   ol = __shfl_xor(bl, off, 32);
      int   oi = __shfl_xor(bi, off, 32);
      if (ov < bv || (ov == bv && ol < bl)) { bv = ov; bl = ol; bi = oi; }
    }
    if (lane == 0)  out[r] = bi;
    if (lane == bl) md[ls] = 3.4e38f;
  }
}

// ---------------------------------------------------------------------------
// Kernel 3a: M = Wq^T * Wk  (f32 accumulate), stored bf16 in swizzled
// B-fragment order (k = first index, o = second).
// ---------------------------------------------------------------------------
__global__ void wqk_kernel(const float* __restrict__ wq, const float* __restrict__ wk,
                           unsigned short* __restrict__ Mswz) {
  int id = blockIdx.x * blockDim.x + threadIdx.x;   // 16384 threads
  int k = id >> 7, o2 = id & 127;
  float acc = 0.0f;
  #pragma unroll 4
  for (int o = 0; o < CH; ++o) acc += wq[o * CH + k] * wk[o * CH + o2];
  Mswz[b_swz_index(k, o2)] = f2bf(acc);
}

// Kernel 3b: transposed bf16 copy into swizzled B order: B[k][o] = w[o][k]
__global__ void wtrans_kernel(const float* __restrict__ w, unsigned short* __restrict__ wswz) {
  int id = blockIdx.x * blockDim.x + threadIdx.x;
  int k = id >> 7, o = id & 127;
  wswz[b_swz_index(k, o)] = f2bf(w[o * CH + k]);
}

// ---------------------------------------------------------------------------
// Kernel 4: WMMA GEMM  out[BN,128] = A_bf16[BN,128] x B_swz[128,128]
// 64 rows/block, 8 waves; each wave keeps its 4 B fragments in VGPRs and
// streams 4 row-tiles of A from LDS -> 16 WMMAs per wave.
// A tile staged with vectorized b128 copies.
// ---------------------------------------------------------------------------
__global__ void gemm_rowtile_kernel(const unsigned short* __restrict__ A,
                                    const unsigned short* __restrict__ Bs,
                                    float* __restrict__ out) {
  __shared__ __align__(16) unsigned short Atile[64 * CH];   // 16 KB
  int tid = threadIdx.x, lane = tid & 31, wave = tid >> 5;
  int r0 = blockIdx.x * 64;
  // prefetch this wave's B fragments into cache while staging A
  __builtin_prefetch(Bs + ((wave * 4) * 32 << 4), 0, 3);
  {
    const uint4* src = reinterpret_cast<const uint4*>(A + (size_t)r0 * CH);
    uint4* dst = reinterpret_cast<uint4*>(Atile);
    #pragma unroll
    for (int i = tid; i < 64 * CH / 8; i += 256) dst[i] = src[i];
  }
  __syncthreads();
  int hi = lane >> 4, col = lane & 15, co = wave * 16;

  v16bf b0 = load_b_frag_swz(Bs, lane, 0, wave);
  v16bf b1 = load_b_frag_swz(Bs, lane, 1, wave);
  v16bf b2 = load_b_frag_swz(Bs, lane, 2, wave);
  v16bf b3 = load_b_frag_swz(Bs, lane, 3, wave);

  #pragma unroll
  for (int rt = 0; rt < 4; ++rt) {
    const unsigned short* at = Atile + rt * 16 * CH;
    v8f acc = {};
    acc = __builtin_amdgcn_wmma_f32_16x16x32_bf16(false, load_a_frag(at, lane, 0), false, b0,
                                                  (short)0, acc, false, false);
    acc = __builtin_amdgcn_wmma_f32_16x16x32_bf16(false, load_a_frag(at, lane, 1), false, b1,
                                                  (short)0, acc, false, false);
    acc = __builtin_amdgcn_wmma_f32_16x16x32_bf16(false, load_a_frag(at, lane, 2), false, b2,
                                                  (short)0, acc, false, false);
    acc = __builtin_amdgcn_wmma_f32_16x16x32_bf16(false, load_a_frag(at, lane, 3), false, b3,
                                                  (short)0, acc, false, false);
    #pragma unroll
    for (int v = 0; v < 8; ++v)
      out[(size_t)(r0 + rt * 16 + v + hi * 8) * CH + co + col] = acc[v];
  }
}

// ---------------------------------------------------------------------------
// Kernel 5: neighbor softmax + aggregation.  One wave per query (2/block).
// Gathers 32 neighbor rows (L2-resident) into LDS while accumulating logits,
// wave32 softmax via shuffles, then attn-weighted channel sum -> s.
// ---------------------------------------------------------------------------
__global__ void attn_aggregate_kernel(const int* __restrict__ idx,
                                      const float* __restrict__ qm,
                                      const float* __restrict__ h,
                                      float* __restrict__ s_f32,
                                      unsigned short* __restrict__ s_b16) {
  __shared__ __align__(16) float gstash[2][KNN_K][CH];   // 32 KB
  __shared__ __align__(16) float qstash[2][CH];
  int tid = threadIdx.x, lane = tid & 31, wave = tid >> 5;
  int q = blockIdx.x * 2 + wave;
  int b = q >> 12;
  #pragma unroll
  for (int t = 0; t < 4; ++t)
    qstash[wave][lane * 4 + t] = qm[(size_t)q * CH + lane * 4 + t];
  __syncthreads();
  int nb = idx[(size_t)q * KNN_K + lane];
  const float* grow = h + ((size_t)(b << 12) + nb) * CH;
  __builtin_prefetch(grow, 0, 3);
  float logit = 0.0f;
  #pragma unroll 8
  for (int c = 0; c < CH; c += 4) {
    float4 g4 = *(const float4*)(grow + c);
    *(float4*)&gstash[wave][lane][c] = g4;
    logit += g4.x * qstash[wave][c] + g4.y * qstash[wave][c + 1]
           + g4.z * qstash[wave][c + 2] + g4.w * qstash[wave][c + 3];
  }
  logit *= 0.08838834764831845f;                 // 1/sqrt(128)
  float m = logit;
  #pragma unroll
  for (int off = 16; off > 0; off >>= 1) m = fmaxf(m, __shfl_xor(m, off, 32));
  float e = __expf(logit - m);
  float ssum = e;
  #pragma unroll
  for (int off = 16; off > 0; off >>= 1) ssum += __shfl_xor(ssum, off, 32);
  float attn = e / ssum;
  __syncthreads();
  float4 acc = make_float4(0.f, 0.f, 0.f, 0.f);
  #pragma unroll 4
  for (int j = 0; j < KNN_K; ++j) {
    float a = __shfl(attn, j, 32);
    float4 g4 = *(const float4*)&gstash[wave][j][lane * 4];
    acc.x += a * g4.x; acc.y += a * g4.y; acc.z += a * g4.z; acc.w += a * g4.w;
  }
  size_t o0 = (size_t)q * CH + lane * 4;
  *(float4*)(s_f32 + o0) = acc;
  uint2 pk;
  pk.x = (unsigned)f2bf(acc.x) | ((unsigned)f2bf(acc.y) << 16);
  pk.y = (unsigned)f2bf(acc.z) | ((unsigned)f2bf(acc.w) << 16);
  *(uint2*)(s_b16 + o0) = pk;
}

// ---------------------------------------------------------------------------
// Kernel 6: fused  u = s*Wv^T ; t = h - u ; w = t*Wff^T ; out = h + relu(bn(w))
// 32 rows/block; each wave keeps 8 B fragments (Wv + Wff) in VGPRs and does
// 2 row-tiles per GEMM -> 16 WMMAs per wave.  Wff fragments prefetched while
// GEMM1 runs.
// ---------------------------------------------------------------------------
__global__ void fused_vff_kernel(const unsigned short* __restrict__ Sb,
                                 const float* __restrict__ Hf,
                                 const unsigned short* __restrict__ WvS,
                                 const unsigned short* __restrict__ WfS,
                                 const float* __restrict__ g, const float* __restrict__ be,
                                 const float* __restrict__ me, const float* __restrict__ va,
                                 float* __restrict__ out_f32,
                                 unsigned short* __restrict__ out_b16) {
  __shared__ __align__(16) unsigned short Stile[32 * CH];   // 8 KB
  __shared__ __align__(16) unsigned short Ttile[32 * CH];   // 8 KB
  int tid = threadIdx.x, lane = tid & 31, wave = tid >> 5;
  int r0 = blockIdx.x * 32;
  // prefetch second-GEMM weights early; they are needed only after barrier 2
  __builtin_prefetch(WfS + ((wave * 4) * 32 << 4), 0, 3);
  {
    const uint4* src = reinterpret_cast<const uint4*>(Sb + (size_t)r0 * CH);
    uint4* dst = reinterpret_cast<uint4*>(Stile);
    #pragma unroll
    for (int i = tid; i < 32 * CH / 8; i += 256) dst[i] = src[i];
  }
  __syncthreads();
  int hi = lane >> 4, col = lane & 15, co = wave * 16;

  v16bf v0 = load_b_frag_swz(WvS, lane, 0, wave);
  v16bf v1 = load_b_frag_swz(WvS, lane, 1, wave);
  v16bf v2 = load_b_frag_swz(WvS, lane, 2, wave);
  v16bf v3 = load_b_frag_swz(WvS, lane, 3, wave);

  #pragma unroll
  for (int rt = 0; rt < 2; ++rt) {
    const unsigned short* st = Stile + rt * 16 * CH;
    v8f acc = {};
    acc = __builtin_amdgcn_wmma_f32_16x16x32_bf16(false, load_a_frag(st, lane, 0), false, v0,
                                                  (short)0, acc, false, false);
    acc = __builtin_amdgcn_wmma_f32_16x16x32_bf16(false, load_a_frag(st, lane, 1), false, v1,
                                                  (short)0, acc, false, false);
    acc = __builtin_amdgcn_wmma_f32_16x16x32_bf16(false, load_a_frag(st, lane, 2), false, v2,
                                                  (short)0, acc, false, false);
    acc = __builtin_amdgcn_wmma_f32_16x16x32_bf16(false, load_a_frag(st, lane, 3), false, v3,
                                                  (short)0, acc, false, false);
    #pragma unroll
    for (int v = 0; v < 8; ++v) {
      int row = rt * 16 + v + hi * 8;
      float t = Hf[(size_t)(r0 + row) * CH + co + col] - acc[v];
      Ttile[row * CH + co + col] = f2bf(t);
    }
  }
  __syncthreads();

  v16bf f0 = load_b_frag_swz(WfS, lane, 0, wave);
  v16bf f1 = load_b_frag_swz(WfS, lane, 1, wave);
  v16bf f2 = load_b_frag_swz(WfS, lane, 2, wave);
  v16bf f3 = load_b_frag_swz(WfS, lane, 3, wave);

  float sc = g[co + col] * rsqrtf(va[co + col] + EPS_BN);
  float bb = be[co + col] - me[co + col] * sc;

  #pragma unroll
  for (int rt = 0; rt < 2; ++rt) {
    const unsigned short* tt = Ttile + rt * 16 * CH;
    v8f acc2 = {};
    acc2 = __builtin_amdgcn_wmma_f32_16x16x32_bf16(false, load_a_frag(tt, lane, 0), false, f0,
                                                   (short)0, acc2, false, false);
    acc2 = __builtin_amdgcn_wmma_f32_16x16x32_bf16(false, load_a_frag(tt, lane, 1), false, f1,
                                                   (short)0, acc2, false, false);
    acc2 = __builtin_amdgcn_wmma_f32_16x16x32_bf16(false, load_a_frag(tt, lane, 2), false, f2,
                                                   (short)0, acc2, false, false);
    acc2 = __builtin_amdgcn_wmma_f32_16x16x32_bf16(false, load_a_frag(tt, lane, 3), false, f3,
                                                   (short)0, acc2, false, false);
    #pragma unroll
    for (int v = 0; v < 8; ++v) {
      int row = rt * 16 + v + hi * 8, o = co + col;
      float w = fmaxf(acc2[v] * sc + bb, 0.0f);
      float res = Hf[(size_t)(r0 + row) * CH + o] + w;
      out_f32[(size_t)(r0 + row) * CH + o] = res;
      if (out_b16) out_b16[(size_t)(r0 + row) * CH + o] = f2bf(res);
    }
  }
}

// ---------------------------------------------------------------------------
extern "C" void kernel_launch(void* const* d_in, const int* in_sizes, int n_in,
                              void* d_out, int out_size, void* d_ws, size_t ws_size,
                              hipStream_t stream) {
  // setup_inputs() dict order:
  const float* x   = (const float*)d_in[0];
  const float* W1  = (const float*)d_in[1];
  const float* W2  = (const float*)d_in[2];
  const float *bn1g = (const float*)d_in[3],  *bn1b = (const float*)d_in[4];
  const float *bn1m = (const float*)d_in[5],  *bn1v = (const float*)d_in[6];
  const float *bn2g = (const float*)d_in[7],  *bn2b = (const float*)d_in[8];
  const float *bn2m = (const float*)d_in[9],  *bn2v = (const float*)d_in[10];
  const float *c1g  = (const float*)d_in[11], *c1b  = (const float*)d_in[12];
  const float *c1m  = (const float*)d_in[13], *c1v  = (const float*)d_in[14];
  const float *c2g  = (const float*)d_in[15], *c2b  = (const float*)d_in[16];
  const float *c2m  = (const float*)d_in[17], *c2v  = (const float*)d_in[18];
  const float *wq1 = (const float*)d_in[19], *wk1 = (const float*)d_in[20];
  const float *wv1 = (const float*)d_in[21], *wf1 = (const float*)d_in[22];
  const float *wq2 = (const float*)d_in[23], *wk2 = (const float*)d_in[24];
  const float *wv2 = (const float*)d_in[25], *wf2 = (const float*)d_in[26];

  char* ws = (char*)d_ws;
  const size_t BF = (size_t)BN * CH * sizeof(float);          // 8 MB
  const size_t BB = (size_t)BN * CH * sizeof(unsigned short); // 4 MB
  const size_t BI = (size_t)BN * KNN_K * sizeof(int);         // 2 MB
  const size_t BW = (size_t)CH * CH * sizeof(unsigned short); // 32 KB
  size_t off = 0;
  float*          h0f = (float*)(ws + off);          off += BF;
  unsigned short* h0b = (unsigned short*)(ws + off); off += BB;
  float*          h1f = (float*)(ws + off);          off += BF;
  unsigned short* h1b = (unsigned short*)(ws + off); off += BB;
  float*          qmb = (float*)(ws + off);          off += BF;
  float*          sf  = (float*)(ws + off);          off += BF;
  unsigned short* sb  = (unsigned short*)(ws + off); off += BB;
  int*            idx = (int*)(ws + off);            off += BI;
  unsigned short* M1   = (unsigned short*)(ws + off); off += BW;
  unsigned short* M2   = (unsigned short*)(ws + off); off += BW;
  unsigned short* WvS1 = (unsigned short*)(ws + off); off += BW;
  unsigned short* WvS2 = (unsigned short*)(ws + off); off += BW;
  unsigned short* WfS1 = (unsigned short*)(ws + off); off += BW;
  unsigned short* WfS2 = (unsigned short*)(ws + off); off += BW;

  // 1. MLP 3->64->128 (+BN,ReLU)
  mlp_kernel<<<BN / 8, 256, 0, stream>>>(x, W1, W2, bn1g, bn1b, bn1m, bn1v,
                                         bn2g, bn2b, bn2m, bn2v, h0f, h0b);
  // 2. 32-NN over xyz (shared by both attention stages)
  knn_kernel<<<BN / 4, 128, 0, stream>>>(x, idx);
  // 3. weight prep (bf16, WMMA-fragment-swizzled B layout)
  wqk_kernel<<<64, 256, 0, stream>>>(wq1, wk1, M1);
  wqk_kernel<<<64, 256, 0, stream>>>(wq2, wk2, M2);
  wtrans_kernel<<<64, 256, 0, stream>>>(wv1, WvS1);
  wtrans_kernel<<<64, 256, 0, stream>>>(wv2, WvS2);
  wtrans_kernel<<<64, 256, 0, stream>>>(wf1, WfS1);
  wtrans_kernel<<<64, 256, 0, stream>>>(wf2, WfS2);

  // ---- cross-attention block 1 ----
  gemm_rowtile_kernel<<<BN / 64, 256, 0, stream>>>(h0b, M1, qmb);       // qm = h*M1 (WMMA)
  attn_aggregate_kernel<<<BN / 2, 64, 0, stream>>>(idx, qmb, h0f, sf, sb);
  fused_vff_kernel<<<BN / 32, 256, 0, stream>>>(sb, h0f, WvS1, WfS1,
                                                c1g, c1b, c1m, c1v, h1f, h1b);
  // ---- cross-attention block 2 (writes final output) ----
  gemm_rowtile_kernel<<<BN / 64, 256, 0, stream>>>(h1b, M2, qmb);       // qm = h*M2 (WMMA)
  attn_aggregate_kernel<<<BN / 2, 64, 0, stream>>>(idx, qmb, h1f, sf, sb);
  fused_vff_kernel<<<BN / 32, 256, 0, stream>>>(sb, h1f, WvS2, WfS2,
                                                c2g, c2b, c2m, c2v,
                                                (float*)d_out, (unsigned short*)nullptr);
  (void)in_sizes; (void)n_in; (void)out_size; (void)ws_size;
}
